// DifferentiablePathfinder_5755256177113
// MI455X (gfx1250) — matrix operand order repaired and checked by reading, not compiled, not run on hardware
//
#include <hip/hip_runtime.h>
#include <hip/hip_bf16.h>

typedef __attribute__((ext_vector_type(16))) _Float16 v16h;
typedef __attribute__((ext_vector_type(8)))  _Float16 v8h;
typedef __attribute__((ext_vector_type(8)))  float    v8f;

#define TAU      0.1f
#define INV_TAU  10.0f
#define INF_D    100.0f
#define ITERS    20

// ---------------------------------------------------------------------------
// Kernel 0 (once): Et[j][i] = adj[i][j] > 0 ? exp(-w[i][j]/tau) : 0   (f16)
// 32x32 LDS-tiled transpose so both the fp32 reads and f16 writes coalesce.
// adj/w are 512 MB of one-shot data: read non-temporally (TH=NT) so they do
// not evict Et from the 192 MB L2 — Et (128 MB, f16) stays L2-resident for
// the 20 GEMV sweeps, so only iteration 1 touches HBM for it.
// ---------------------------------------------------------------------------
__global__ void __launch_bounds__(256)
dpf_build_et(const int* __restrict__ adj, const float* __restrict__ w,
             _Float16* __restrict__ Et, int n) {
  __shared__ _Float16 tile[32][33];   // +1 pad: kill bank conflicts
  const int i0 = blockIdx.y * 32;
  const int j0 = blockIdx.x * 32;
  const int tx = threadIdx.x;         // 0..31
  const int ty = threadIdx.y;         // 0..7
#pragma unroll
  for (int q = 0; q < 4; ++q) {
    const int r = ty + q * 8;
    const size_t idx = (size_t)(i0 + r) * (size_t)n + (size_t)(j0 + tx);
    const int   a = __builtin_nontemporal_load(&adj[idx]);
    const float x = __builtin_nontemporal_load(&w[idx]);
    const float e = (a > 0) ? __expf(-INV_TAU * x) : 0.0f;
    tile[r][tx] = (_Float16)e;        // tile[a][b] = E(i0+a, j0+b)
  }
  __syncthreads();
#pragma unroll
  for (int q = 0; q < 4; ++q) {
    const int c = ty + q * 8;
    Et[(size_t)(j0 + c) * (size_t)n + (size_t)(i0 + tx)] = tile[tx][c];
  }
}

// ---------------------------------------------------------------------------
// Init: d[j] = source ? 0 : INF
// ---------------------------------------------------------------------------
__global__ void dpf_init_d(const int* __restrict__ mask, float* __restrict__ d, int n) {
  const int j = blockIdx.x * blockDim.x + threadIdx.x;
  if (j < n) d[j] = (mask[j] > 0) ? 0.0f : INF_D;
}

// ---------------------------------------------------------------------------
// Kernel 1 (per iter): dmin = min_j d[j];  a16[i] = f16(exp(-(d[i]-dmin)/tau))
// Single 1024-thread block (n = 8192 -> 8 elements/thread). Cheap.
// ---------------------------------------------------------------------------
__global__ void __launch_bounds__(1024)
dpf_prep_a(const float* __restrict__ d, _Float16* __restrict__ a16,
           float* __restrict__ dmin_out, int n) {
  __shared__ float red[1024];
  const int t = threadIdx.x;
  float m = 3.4e38f;
  for (int i = t; i < n; i += 1024) m = fminf(m, d[i]);
  red[t] = m;
  __syncthreads();
  for (int s = 512; s > 0; s >>= 1) {
    if (t < s) red[t] = fminf(red[t], red[t + s]);
    __syncthreads();
  }
  const float dmin = red[0];
  if (t == 0) *dmin_out = dmin;
  for (int i = t; i < n; i += 1024)
    a16[i] = (_Float16)__expf(-INV_TAU * (d[i] - dmin));
}

// ---------------------------------------------------------------------------
// Kernel 2 (per iter): WMMA GEMV, K-unrolled x4.
//   D(16x16) = A(16x32 f16) x B(32x16 f16) + C, accumulated over K = 8192.
//   A row M = output column j0+M = Et[j0+M][k...]   (contiguous in memory)
//   B      = a[k...] replicated across all 16 columns -> all D columns equal.
// A-operand f16 layout (ISA 7.12.2): lanes 0-15 hold K = k0+{0..7, 16..23},
// lanes 16-31 hold K = k0+{8..15, 24..31}  (two 16B loads per lane).
// B-operand: same split as A, lanes 0-15 hold K = k0+{0..7, 16..23}, lanes 16-31 K = k0+{8..15, 24..31}.
// C/D: VGPR r -> M=r (lanes 0-15) / M=8+r (lanes 16-31); N = lane%16.
// 8 waves/block split K; partials reduced through LDS; finalize in-block.
// ---------------------------------------------------------------------------
__global__ void __launch_bounds__(256)
dpf_gemv_wmma(const _Float16* __restrict__ Et, const _Float16* __restrict__ a16,
              const float* __restrict__ dmin_p, float* __restrict__ d, int n) {
  __shared__ float partial[8][16];
  const int lane = threadIdx.x & 31;
  const int wave = threadIdx.x >> 5;
  const int j0   = blockIdx.x * 16;
  const int row  = j0 + (lane & 15);
  const int half = lane >> 4;              // which 16-lane group
  const int kChunk = n >> 3;               // 8192 / 8 waves = 1024
  const int kBegin = wave * kChunk;
  const int kEnd   = kBegin + kChunk;

  const _Float16* __restrict__ arow = Et + (size_t)row * (size_t)n;

  v8f c = {};
  for (int k0 = kBegin; k0 < kEnd; k0 += 128) {   // 4 WMMAs per outer iter
    v16h A[4], B[4];
#pragma unroll
    for (int u = 0; u < 4; ++u) {
      const int k = k0 + u * 32;
      // A: this lane's row, K = k + half*8 + {0..7} and k + 16 + half*8 + {0..7}
      v8h alo = *(const v8h*)(arow + k + half * 8);
      v8h ahi = *(const v8h*)(arow + k + 16 + half * 8);
      A[u] = __builtin_shufflevector(alo, ahi,
                                     0, 1, 2, 3, 4, 5, 6, 7,
                                     8, 9, 10, 11, 12, 13, 14, 15);
      // B: broadcast vector a, K = k + half*8 + {0..7} and k + 16 + half*8 + {0..7}
      v8h blo = *(const v8h*)(a16 + k + half * 8);
      v8h bhi = *(const v8h*)(a16 + k + 16 + half * 8);
      B[u] = __builtin_shufflevector(blo, bhi,
                                     0, 1, 2, 3, 4, 5, 6, 7,
                                     8, 9, 10, 11, 12, 13, 14, 15);
    }
    // stream-ahead hint for the A row (speculative; drops silently if OOB)
    __builtin_prefetch((const void*)(arow + k0 + 1024), 0, 0);
#pragma unroll
    for (int u = 0; u < 4; ++u) {
      c = __builtin_amdgcn_wmma_f32_16x16x32_f16(
              /*neg_a=*/false, A[u], /*neg_b=*/false, B[u],
              /*c_mod=*/(short)0, c, /*reuse_a=*/false, /*reuse_b=*/false);
    }
  }

  // All D columns identical: pull y partials off lanes 0 (M=0..7) and 16 (M=8..15).
  if (lane == 0) {
#pragma unroll
    for (int m = 0; m < 8; ++m) partial[wave][m] = c[m];
  } else if (lane == 16) {
#pragma unroll
    for (int m = 0; m < 8; ++m) partial[wave][8 + m] = c[m];
  }
  __syncthreads();

  if (threadIdx.x < 16) {
    float y = 0.0f;
#pragma unroll
    for (int wv = 0; wv < 8; ++wv) y += partial[wv][threadIdx.x];
    const float dmin = *dmin_p;
    const float nd = dmin - TAU * __logf(y);   // y==0 -> +inf -> keeps old d
    const int j = j0 + threadIdx.x;
    d[j] = fminf(d[j], nd);
  }
}

// ---------------------------------------------------------------------------
extern "C" void kernel_launch(void* const* d_in, const int* in_sizes, int n_in,
                              void* d_out, int out_size, void* d_ws, size_t ws_size,
                              hipStream_t stream) {
  const int*   adj  = (const int*)d_in[0];
  const float* w    = (const float*)d_in[1];
  const int*   mask = (const int*)d_in[2];
  float*       dist = (float*)d_out;
  const int n = in_sizes[2];               // 8192

  // workspace layout: [ Et f16 : n*n*2 B ][ a16 : n*2 B ][ dmin : 4 B ]
  char* ws = (char*)d_ws;
  _Float16* Et   = (_Float16*)ws;
  const size_t etBytes = (size_t)n * (size_t)n * sizeof(_Float16);
  _Float16* a16  = (_Float16*)(ws + etBytes);
  float*    dmin = (float*)(ws + etBytes + (size_t)n * sizeof(_Float16));

  dim3 tb(32, 8);
  dim3 tg(n / 32, n / 32);
  dpf_build_et<<<tg, tb, 0, stream>>>(adj, w, Et, n);
  dpf_init_d<<<(n + 255) / 256, 256, 0, stream>>>(mask, dist, n);

  for (int it = 0; it < ITERS; ++it) {
    dpf_prep_a<<<1, 1024, 0, stream>>>(dist, a16, dmin, n);
    dpf_gemv_wmma<<<n / 16, 256, 0, stream>>>(Et, a16, dmin, dist, n);
  }
}
